// BiLSTMCRF_75153337745459
// MI455X (gfx1250) — matrix operand order repaired and checked by reading, not compiled, not run on hardware
//
#include <hip/hip_runtime.h>
#include <hip/hip_bf16.h>

// ---------------------------------------------------------------------------
// BiLSTM-CRF forward loss for MI455X (gfx1250).
//
// - Bulk time-parallel input GEMM x@Wih^T via v_wmma_f32_16x16x32_f16,
//   with the weight tile brought in by async global->LDS copies
//   (global_load_async_to_lds_b128 + s_wait_asynccnt) and the embedding
//   gather done cooperatively into LDS.
// - Pre-gates stored TRANSPOSED per timestep ([t][n][m]) so the serial LSTM
//   kernel loads each WMMA accumulator fragment as 2 contiguous b128 loads.
// - Persistent-WGP recurrence: 1 workgroup/direction, h double-buffered in
//   LDS (f16), Whh B-fragments register-resident, 1 barrier/step, coalesced
//   cooperative h writeback.
// - CRF is scalar f32 (K=9), one thread per batch element.
// ---------------------------------------------------------------------------

typedef _Float16 half_t;
typedef _Float16 v16h __attribute__((ext_vector_type(16)));
typedef float    v8f  __attribute__((ext_vector_type(8)));

#define S_LEN  512
#define BATCH  128
#define EMB    100
#define EPAD   128          // E padded to WMMA K multiple (zeros)
#define HID    64           // per-direction hidden
#define GATES  256          // 4*HID, PyTorch gate order i,f,g,o
#define NTAG   9            // T-1
#define TPAD   16           // tag dim padded for WMMA N
#define ROWS   (S_LEN*BATCH) // 65536 flattened (t,b) rows

static __device__ __forceinline__ float sigmoidf_(float x) {
  return 1.0f / (1.0f + __expf(-x));
}

// Async 16-byte global->LDS copy (CDNA5 VGLOBAL async op, ASYNCcnt-tracked)
static __device__ __forceinline__ void async_copy_b128(unsigned lds_off,
                                                       const void* gaddr) {
  asm volatile("global_load_async_to_lds_b128 %0, %1, off"
               :: "v"(lds_off), "v"((unsigned long long)(size_t)gaddr)
               : "memory");
}
static __device__ __forceinline__ void wait_async0() {
  asm volatile("s_wait_asynccnt 0x0" ::: "memory");
}

// --- WMMA fragment loaders, per ISA 7.12.2 VGPR layouts (wave32) -----------
// A (16x32 f16): lane L -> m = L&15, hi = L>=16; element 2v+p -> k =
//   (v<4?0:16) + hi*8 + (v&3)*2 + p
static __device__ __forceinline__
v16h load_fragA(const half_t* tile, int ldk, int mBase, int kBase, int lane) {
  const int hi = (lane >> 4) & 1;
  const int m  = mBase + (lane & 15);
  v16h a;
#pragma unroll
  for (int v = 0; v < 8; ++v)
#pragma unroll
    for (int p = 0; p < 2; ++p) {
      const int k = kBase + ((v < 4) ? 0 : 16) + hi * 8 + (v & 3) * 2 + p;
      a[v * 2 + p] = tile[(size_t)m * ldk + k];
    }
  return a;
}

// B (32x16 f16, tile stored [k][n], leading dim ldn): lane L -> n = L&15,
// hi = L>=16; element 2v+p -> k = (v<4?0:16) + hi*8 + (v&3)*2 + p
static __device__ __forceinline__
v16h load_fragB_kn(const half_t* tile, int ldn, int kBase, int nBase, int lane) {
  const int hi = (lane >> 4) & 1;
  const int n = nBase + (lane & 15);
  v16h b;
#pragma unroll
  for (int v = 0; v < 8; ++v)
#pragma unroll
    for (int p = 0; p < 2; ++p) {
      const int k = kBase + ((v < 4) ? 0 : 16) + hi * 8 + (v & 3) * 2 + p;
      b[v * 2 + p] = tile[(size_t)k * ldn + n];
    }
  return b;
}

// B fragment straight from a row-major weight W[n][k] (B[k][n] = W[n][k])
static __device__ __forceinline__
v16h load_fragB_wt(const half_t* W, int ldk, int kBase, int nBase, int lane) {
  const int hi = (lane >> 4) & 1;
  const int n = nBase + (lane & 15);
  v16h b;
#pragma unroll
  for (int v = 0; v < 8; ++v)
#pragma unroll
    for (int p = 0; p < 2; ++p) {
      const int k = kBase + ((v < 4) ? 0 : 16) + hi * 8 + (v & 3) * 2 + p;
      b[v * 2 + p] = W[(size_t)n * ldk + k];
    }
  return b;
}

// ---------------------------------------------------------------------------
// Kernel 0: weights -> f16. Wih stored TRANSPOSED [dir][k=128][n=256] so the
// GEMM B-tile is a straight async-copyable slab. Whh/Wout row-major, padded.
// ---------------------------------------------------------------------------
__global__ __launch_bounds__(256)
void convert_weights(const float* __restrict__ Wih_f, const float* __restrict__ Whh_f,
                     const float* __restrict__ Wih_b, const float* __restrict__ Whh_b,
                     const float* __restrict__ W_out,
                     half_t* __restrict__ Wih_hT,  // [2][128][256]
                     half_t* __restrict__ Whh_h,   // [2][256][64]
                     half_t* __restrict__ Wout_h)  // [16][128]
{
  const int i = blockIdx.x * blockDim.x + threadIdx.x;
  if (i < 2 * EPAD * GATES) {
    const int d = i / (EPAD * GATES), k = (i / GATES) % EPAD, n = i % GATES;
    const float* W = d ? Wih_b : Wih_f;
    Wih_hT[i] = (half_t)((k < EMB) ? W[n * EMB + k] : 0.0f);
  }
  if (i < 2 * GATES * HID) {
    const int d = i / (GATES * HID), r = (i / HID) % GATES, k = i % HID;
    const float* W = d ? Whh_b : Whh_f;
    Whh_h[i] = (half_t)W[r * HID + k];
  }
  if (i < TPAD * EPAD) {
    const int r = i / EPAD, k = i % EPAD;
    Wout_h[i] = (half_t)((r < 10) ? W_out[r * EPAD + k] : 0.0f);
  }
}

// ---------------------------------------------------------------------------
// Kernel 1: fused embedding gather + pre-gate GEMM.
//   preT[dir][t][n][b] = bias[n] + sum_k emb[tok(t,b)][k] * Wih[n][k]
// Block = 128 threads (4 waves), tile 64(M=rows of one t) x 64(N), K=128.
// B tile streamed global->LDS with async b128 copies; A gathered + f16-cast.
// Output stored timestep-transposed so the LSTM reads contiguous fragments.
// ---------------------------------------------------------------------------
__global__ __launch_bounds__(128)
void pregate_gemm(const int* __restrict__ inputs,    // [B][S]
                  const float* __restrict__ emb,     // [VOCAB][100]
                  const half_t* __restrict__ Wih_hT, // [2][128][256]
                  const float* __restrict__ b_f, const float* __restrict__ b_b,
                  float* __restrict__ pre)           // [2][S][256][128]
{
  __shared__ half_t Atile[64][EPAD];   // 16 KB, gathered embeddings (f16)
  __shared__ half_t Btile[EPAD][64];   // 16 KB, Wih^T slice  [k][n]

  const int dir = blockIdx.z;
  const int r0  = blockIdx.y * 64;     // 64 rows within one timestep
  const int n0  = blockIdx.x * 64;
  const int t_blk = r0 >> 7;           // timestep of this block
  const int bBase = r0 & 127;          // 0 or 64
  const int tid = threadIdx.x;
  const half_t* Wsrc = Wih_hT + (size_t)dir * EPAD * GATES;

  // B tile: pure slab copy, async global->LDS (128 rows x 128 B; one row
  // of 8 x b128 chunks per thread, no VGPR staging, overlaps the A gather)
  {
    const unsigned lbase = (unsigned)(size_t)&Btile[0][0];
    const half_t* grow = Wsrc + (size_t)tid * GATES + n0;
#pragma unroll
    for (int it = 0; it < 8; ++it)
      async_copy_b128(lbase + tid * 128 + it * 16, grow + it * 8);
  }

  // A tile: gather embedding rows, f32 -> f16, 64 halves per thread
#pragma unroll 4
  for (int it = 0; it < 64; ++it) {
    const int flat = tid * 64 + it;
    const int rl = flat >> 7, k = flat & 127;
    const int r = r0 + rl, t = r >> 7, b = r & 127;
    const int tok = inputs[b * S_LEN + t];
    Atile[rl][k] = (half_t)((k < EMB) ? emb[(size_t)tok * EMB + k] : 0.0f);
  }
  wait_async0();
  __syncthreads();

  const int wave = tid >> 5, lane = tid & 31;
  const int hi = (lane >> 4) & 1;
  const int mBase = wave * 16;
  const float* bias = dir ? b_b : b_f;

  v16h afr[4];
#pragma unroll
  for (int ks = 0; ks < 4; ++ks)
    afr[ks] = load_fragA(&Atile[0][0], EPAD, mBase, ks * 32, lane);

  float* dstT = pre + ((size_t)dir * S_LEN + t_blk) * GATES * BATCH;
#pragma unroll
  for (int nt = 0; nt < 4; ++nt) {
    const int n = n0 + nt * 16 + (lane & 15);
    const float bv = bias[n];
    v8f c;
#pragma unroll
    for (int e = 0; e < 8; ++e) c[e] = bv;    // bias folded into C init
#pragma unroll
    for (int ks = 0; ks < 4; ++ks) {
      v16h bfr = load_fragB_kn(&Btile[0][0], 64, ks * 32, nt * 16, lane);
      c = __builtin_amdgcn_wmma_f32_16x16x32_f16(false, afr[ks], false, bfr,
                                                 (short)0, c, false, false);
    }
    // transposed store: lane's 8 fragment elements are contiguous in m(=b)
    float4* dp = (float4*)(dstT + (size_t)n * BATCH + bBase + mBase + hi * 8);
    dp[0] = float4{c[0], c[1], c[2], c[3]};
    dp[1] = float4{c[4], c[5], c[6], c[7]};
  }
}

// ---------------------------------------------------------------------------
// Kernel 2: persistent LSTM recurrence. grid.x = 2 (dir), 256 threads/8 waves.
// h state double-buffered in LDS (f16); Whh WMMA B-fragments preloaded into
// VGPRs (loop invariant). Per step: 8 waves x 4 (mt,nt0) combos x 4 gates x
// 2 k-steps = 256 WMMAs -> full 128x256 gate GEMM; fused sigmoid/tanh cell
// update on accumulator fragments; one barrier/step; coalesced h writeback.
// ---------------------------------------------------------------------------
__global__ __launch_bounds__(256)
void lstm_dir_kernel(const float* __restrict__ pre,    // [2][S][256][128]
                     const half_t* __restrict__ Whh_h, // [2][256][64]
                     half_t* __restrict__ h_all)       // [ROWS][128] f16
{
  __shared__ half_t hb[2][BATCH * HID];   // 32 KB double-buffered h (f16)

  const int dir  = blockIdx.x;
  const int rev  = dir;                   // dir 1 scans backward
  const int tid  = threadIdx.x;
  const int wave = tid >> 5, lane = tid & 31;
  const int hi   = (lane >> 4) & 1;
  const int nt0  = wave & 3;              // hidden-unit n-tile (j = nt0*16+..)
  const int mhalf = wave >> 2;            // batch half

  const float*  pre_d = pre   + (size_t)dir * S_LEN * GATES * BATCH;
  const half_t* Whh_d = Whh_h + (size_t)dir * GATES * HID;

  // Preload loop-invariant Whh B-fragments: [gate][kstep]
  v16h bfrag[4][2];
#pragma unroll
  for (int g = 0; g < 4; ++g)
#pragma unroll
    for (int ks = 0; ks < 2; ++ks)
      bfrag[g][ks] = load_fragB_wt(Whh_d, HID, ks * 32,
                                   g * HID + nt0 * 16, lane);

  float c_reg[4][8];
#pragma unroll
  for (int ci = 0; ci < 4; ++ci)
#pragma unroll
    for (int v = 0; v < 8; ++v) c_reg[ci][v] = 0.0f;

  for (int i = tid; i < BATCH * HID; i += 256) hb[0][i] = (half_t)0.0f;
  __syncthreads();

  for (int st = 0; st < S_LEN; ++st) {
    const int t   = rev ? (S_LEN - 1 - st) : st;
    const int cur = st & 1, nxt = cur ^ 1;
    const float* pt = pre_d + (size_t)t * GATES * BATCH;  // [n][m] slice

    // prefetch next step's pre-gate slab into the cache hierarchy
    if (st + 1 < S_LEN) {
      const int tn = rev ? (S_LEN - 2 - st) : (st + 1);
      __builtin_prefetch(pre_d + (size_t)tn * GATES * BATCH + tid * 128, 0, 1);
    }

#pragma unroll
    for (int ci = 0; ci < 4; ++ci) {
      const int mt = mhalf * 4 + ci;
      const v16h a0 = load_fragA(hb[cur], HID, mt * 16, 0, lane);
      const v16h a1 = load_fragA(hb[cur], HID, mt * 16, 32, lane);

      v8f g4[4];
#pragma unroll
      for (int g = 0; g < 4; ++g) {
        const int n = g * HID + nt0 * 16 + (lane & 15);
        // transposed pre: lane's 8 accumulator inits are contiguous -> 2xb128
        const float4* pp =
            (const float4*)(pt + (size_t)n * BATCH + mt * 16 + hi * 8);
        const float4 p0 = pp[0], p1 = pp[1];
        v8f c;
        c[0] = p0.x; c[1] = p0.y; c[2] = p0.z; c[3] = p0.w;
        c[4] = p1.x; c[5] = p1.y; c[6] = p1.z; c[7] = p1.w;
        c = __builtin_amdgcn_wmma_f32_16x16x32_f16(false, a0, false,
                                                   bfrag[g][0], (short)0, c,
                                                   false, false);
        c = __builtin_amdgcn_wmma_f32_16x16x32_f16(false, a1, false,
                                                   bfrag[g][1], (short)0, c,
                                                   false, false);
        g4[g] = c;
      }

      // fused cell update on accumulator fragments (i,f,g,o aligned per elem)
      const int j = nt0 * 16 + (lane & 15);
#pragma unroll
      for (int v = 0; v < 8; ++v) {
        const float ig = sigmoidf_(g4[0][v]);
        const float fg = sigmoidf_(g4[1][v]);
        const float gg = tanhf(g4[2][v]);
        const float og = sigmoidf_(g4[3][v]);
        const float cc = fg * c_reg[ci][v] + ig * gg;
        c_reg[ci][v] = cc;
        const float h = og * tanhf(cc);
        const int m = mt * 16 + v + hi * 8;
        hb[nxt][m * HID + j] = (half_t)h;       // next-step state (LDS only)
      }
    }
    __syncthreads();   // hb[nxt] complete

    // coalesced cooperative writeback of h(t): LDS -> global, 4 x uint4/thread
    {
      const uint4* src = (const uint4*)hb[nxt];
#pragma unroll
      for (int q = 0; q < 4; ++q) {
        const int idx = tid * 4 + q;            // 1024 x 16B chunks
        const int m = idx >> 3, part = idx & 7; // 8 chunks per batch row
        uint4* dp = (uint4*)(h_all + ((size_t)t * BATCH + m) * EPAD + dir * HID);
        dp[part] = src[idx];
      }
    }
    // no extra barrier: next iteration writes hb[cur] (the other buffer)
  }
}

// ---------------------------------------------------------------------------
// Kernel 3: output projection em = h_all @ W_out^T + b_out (N padded to 16).
// 4 waves/block, one 16x16 output tile per wave, K=128 -> 4 WMMAs.
// ---------------------------------------------------------------------------
__global__ __launch_bounds__(128)
void proj_kernel(const half_t* __restrict__ h_all,  // [ROWS][128] f16
                 const half_t* __restrict__ Wout_h, // [16][128] f16
                 const float* __restrict__ b_out,   // [10]
                 float* __restrict__ em)            // [ROWS][16] f32
{
  const int wave = threadIdx.x >> 5, lane = threadIdx.x & 31;
  const int hi = (lane >> 4) & 1;
  const int r0 = (blockIdx.x * 4 + wave) * 16;
  const int n  = lane & 15;

  v8f c;
  const float bv = (n < 10) ? b_out[n] : 0.0f;
#pragma unroll
  for (int e = 0; e < 8; ++e) c[e] = bv;

#pragma unroll
  for (int ks = 0; ks < 4; ++ks) {
    const v16h a = load_fragA(h_all, EPAD, r0, ks * 32, lane);
    const v16h b = load_fragB_wt(Wout_h, EPAD, ks * 32, 0, lane);
    c = __builtin_amdgcn_wmma_f32_16x16x32_f16(false, a, false, b,
                                               (short)0, c, false, false);
  }
#pragma unroll
  for (int v = 0; v < 8; ++v) {
    const int r = r0 + v + hi * 8;
    em[(size_t)r * TPAD + n] = c[v];
  }
}

// ---------------------------------------------------------------------------
// Kernel 4: linear-chain CRF NLL (mask all-true). One thread per batch;
// alpha[9] register-resident, trans in LDS; block reduction to scalar loss.
// em columns 1..9 correspond to the reference's [:, :, 1:] slice.
// ---------------------------------------------------------------------------
__global__ __launch_bounds__(128)
void crf_kernel(const float* __restrict__ em,    // [ROWS][16]
                const int* __restrict__ tags,    // [B][S]
                const float* __restrict__ start, const float* __restrict__ endv,
                const float* __restrict__ trans, // [9][9]
                float* __restrict__ out)
{
  __shared__ float tr[NTAG * NTAG];
  __shared__ float red[BATCH];
  const int b = threadIdx.x;
  if (b < NTAG * NTAG) tr[b] = trans[b];
  __syncthreads();

  float alpha[NTAG];
  const float* e0 = em + (size_t)b * TPAD;     // t=0 row
  const int tg0 = tags[b * S_LEN] - 1;
  float score = start[tg0] + e0[1 + tg0];
#pragma unroll
  for (int j = 0; j < NTAG; ++j) alpha[j] = start[j] + e0[1 + j];
  int tp = tg0;

  for (int t = 1; t < S_LEN; ++t) {
    const float* et = em + ((size_t)t * BATCH + b) * TPAD;
    const int tg = tags[b * S_LEN + t] - 1;
    score += tr[tp * NTAG + tg] + et[1 + tg];
    tp = tg;
    float na[NTAG];
#pragma unroll
    for (int j = 0; j < NTAG; ++j) {
      float mx = -1e30f;
#pragma unroll
      for (int i = 0; i < NTAG; ++i) mx = fmaxf(mx, alpha[i] + tr[i * NTAG + j]);
      float s = 0.0f;
#pragma unroll
      for (int i = 0; i < NTAG; ++i) s += __expf(alpha[i] + tr[i * NTAG + j] - mx);
      na[j] = mx + __logf(s) + et[1 + j];
    }
#pragma unroll
    for (int j = 0; j < NTAG; ++j) alpha[j] = na[j];
  }
  score += endv[tp];
  float mx = -1e30f;
#pragma unroll
  for (int j = 0; j < NTAG; ++j) mx = fmaxf(mx, alpha[j] + endv[j]);
  float s = 0.0f;
#pragma unroll
  for (int j = 0; j < NTAG; ++j) s += __expf(alpha[j] + endv[j] - mx);
  const float logZ = mx + __logf(s);

  red[b] = logZ - score;                // per-batch -log_likelihood
  __syncthreads();
  for (int off = 64; off > 0; off >>= 1) {
    if (b < off) red[b] += red[b + off];
    __syncthreads();
  }
  if (b == 0) out[0] = red[0] / (float)BATCH;
}

// ---------------------------------------------------------------------------
extern "C" void kernel_launch(void* const* d_in, const int* in_sizes, int n_in,
                              void* d_out, int out_size, void* d_ws, size_t ws_size,
                              hipStream_t stream) {
  (void)in_sizes; (void)n_in; (void)out_size; (void)ws_size;
  const int*   inputs = (const int*)  d_in[0];
  const int*   tags   = (const int*)  d_in[1];
  /* d_in[2] = mask: all-true in this workload, unused */
  const float* emb    = (const float*)d_in[3];
  const float* Wih_f  = (const float*)d_in[4];
  const float* Whh_f  = (const float*)d_in[5];
  const float* b_f    = (const float*)d_in[6];
  const float* Wih_b  = (const float*)d_in[7];
  const float* Whh_b  = (const float*)d_in[8];
  const float* b_b    = (const float*)d_in[9];
  const float* W_out  = (const float*)d_in[10];
  const float* b_out  = (const float*)d_in[11];
  const float* start  = (const float*)d_in[12];
  const float* endv   = (const float*)d_in[13];
  const float* trans  = (const float*)d_in[14];
  float* out = (float*)d_out;

  // workspace layout (256B aligned)
  char* ws = (char*)d_ws;
  size_t o = 0;
  auto take = [&](size_t bytes) { void* p = ws + o; o = (o + bytes + 255) & ~(size_t)255; return p; };
  half_t* Wih_hT = (half_t*)take((size_t)2 * EPAD * GATES * sizeof(half_t));  // 128 KB
  half_t* Whh_h  = (half_t*)take((size_t)2 * GATES * HID  * sizeof(half_t));  //  64 KB
  half_t* Wout_h = (half_t*)take((size_t)TPAD * EPAD      * sizeof(half_t));  //   4 KB
  float*  pre    = (float*) take((size_t)2 * ROWS * GATES * sizeof(float));   // 128 MB
  half_t* h_all  = (half_t*)take((size_t)ROWS * EPAD      * sizeof(half_t));  //  16 MB
  float*  em     = (float*) take((size_t)ROWS * TPAD      * sizeof(float));   //   4 MB

  // 0) weight convert (covers max index 2*128*256 = 65536)
  convert_weights<<<256, 256, 0, stream>>>(Wih_f, Whh_f, Wih_b, Whh_b, W_out,
                                           Wih_hT, Whh_h, Wout_h);
  // 1) fused gather + input GEMM: grid (N tiles, M tiles, dir)
  pregate_gemm<<<dim3(GATES / 64, ROWS / 64, 2), 128, 0, stream>>>(
      inputs, emb, Wih_hT, b_f, b_b, pre);
  // 2) persistent recurrence, one workgroup per direction
  lstm_dir_kernel<<<2, 256, 0, stream>>>(pre, Whh_h, h_all);
  // 3) projection to (padded) tag scores
  proj_kernel<<<ROWS / 64, 128, 0, stream>>>(h_all, Wout_h, b_out, em);
  // 4) CRF NLL + mean reduction
  crf_kernel<<<1, BATCH, 0, stream>>>(em, tags, start, endv, trans, out);
}
